// GRU_4BiRNN_56530359550207
// MI455X (gfx1250) — matrix operand order repaired and checked by reading, not compiled, not run on hardware
//
#include <hip/hip_runtime.h>
#include <math.h>

typedef unsigned int u32;
typedef unsigned short u16;
typedef __attribute__((ext_vector_type(2)))  float  f32x2;
typedef __attribute__((ext_vector_type(4)))  u32    u32x4;
typedef __attribute__((ext_vector_type(16))) __bf16 v16bf;
typedef __attribute__((ext_vector_type(8)))  float  v8f;

struct BfPair { u32x4 lo, hi; };

static constexpr size_t MAT = 512ull * 512ull;          // elems per 512x512 matrix
static constexpr size_t GS  = 512ull * 64ull * 512ull;  // elems per gate in proj [T][B][512]

static __device__ __forceinline__ u32 bf16_1(float x) {
  u32 u = __float_as_uint(x);
  return (u + 0x7fffu + ((u >> 16) & 1u)) >> 16;        // round-to-nearest-even
}
static __device__ __forceinline__ u32 bf16_2(float a, float b) {
  return bf16_1(a) | (bf16_1(b) << 16);
}
// A operand (16x32 bf16): lane holds K = {c..c+7} and {c+16..c+23}: chunks at +0B and +32B
static __device__ __forceinline__ v16bf ldA(const u16* p) {
  BfPair t; t.lo = *(const u32x4*)p; t.hi = *(const u32x4*)(p + 16);
  return __builtin_bit_cast(v16bf, t);
}
// B operand (32x16 bf16): lane holds K = {c..c+7} and {c+16..c+23}: chunks at +0B and +32B
static __device__ __forceinline__ v16bf ldB(const u16* p) {
  BfPair t; t.lo = *(const u32x4*)p; t.hi = *(const u32x4*)(p + 16);
  return __builtin_bit_cast(v16bf, t);
}
static __device__ __forceinline__ v8f wmma_bf(v16bf a, v16bf b, v8f c) {
  return __builtin_amdgcn_wmma_f32_16x16x32_bf16(false, a, false, b, (short)0, c, false, false);
}
static __device__ __forceinline__ v8f vzero() {
  v8f z = {0.f, 0.f, 0.f, 0.f, 0.f, 0.f, 0.f, 0.f};
  return z;
}
static __device__ __forceinline__ float sigm(float x) { return 1.0f / (1.0f + expf(-x)); }

// ---------------- Phase 0: transpose + convert the 6 weight matrices to bf16 ----------------
// Wt[g][n][k] = W_g[k][n], Ut[g][n][k] = U_g[k][n]   (g: 0=z, 1=r, 2=h)
__global__ void __launch_bounds__(256) k_prep(
    const float* __restrict__ Wz, const float* __restrict__ Wr, const float* __restrict__ Wh,
    const float* __restrict__ Uz, const float* __restrict__ Ur, const float* __restrict__ Uh,
    u16* __restrict__ Wt, u16* __restrict__ Ut)
{
  const u32 idx = blockIdx.x * 256u + threadIdx.x;   // exactly 6*512*512 threads launched
  const int mat = idx >> 18;
  const int rem = idx & 262143;
  const int n = rem >> 9;
  const int k = rem & 511;
  const float* src; u16* dst;
  switch (mat) {
    case 0:  src = Wz; dst = Wt;            break;
    case 1:  src = Wr; dst = Wt + MAT;      break;
    case 2:  src = Wh; dst = Wt + 2 * MAT;  break;
    case 3:  src = Uz; dst = Ut;            break;
    case 4:  src = Ur; dst = Ut + MAT;      break;
    default: src = Uh; dst = Ut + 2 * MAT;  break;
  }
  dst[(size_t)n * 512 + k] = (u16)bf16_1(src[(size_t)k * 512 + n]);
}

// ---------------- Phase 1: input projections proj[g][t][b][n] = X[b,t,:]·W_g[:,n] + b_g[n] ----
// Grid: 2048 blocks (one 16-row M-tile of X each) x 256 threads (8 waves).
// The X tile is converted to bf16 in LDS ONCE per block; each wave then runs 4 N-tiles with
// 3 gate accumulators sharing the ds_load_b128 A-operand loads.
__global__ void __launch_bounds__(256) k_proj(
    const float* __restrict__ X, const u16* __restrict__ Wt,
    const float* __restrict__ bz, const float* __restrict__ br, const float* __restrict__ bh,
    float* __restrict__ proj)
{
  __shared__ u16 sXA[16][520];       // bf16 X tile, padded rows -> conflict-free ds_load_b128

  const int lane = threadIdx.x & 31;
  const int wv   = threadIdx.x >> 5;
  const int hf   = lane >> 4;        // half-wave select
  const int l16  = lane & 15;
  const int m0   = blockIdx.x * 16;  // rows m0..m0+15 of X (fixed b, 16 consecutive t)

  // stage + convert: 16 rows x 256 dword-pairs, coalesced f32x2 loads, packed ds_store_b32
  for (int e = threadIdx.x; e < 16 * 256; e += 256) {
    const int row = e >> 8;
    const int c2  = e & 255;                       // dword index within row
    f32x2 x = *(const f32x2*)(X + (size_t)(m0 + row) * 512 + 2 * c2);
    ((u32*)&sXA[row][0])[c2] = bf16_2(x.x, x.y);
  }
  __syncthreads();

  for (int j = 0; j < 4; ++j) {
    const int n0 = (wv * 4 + j) * 16;
    v8f acc0 = vzero(), acc1 = vzero(), acc2 = vzero();
    for (int kb = 0; kb < 16; ++kb) {
      const v16bf A = ldA(&sXA[l16][kb * 32 + hf * 8]);
      const size_t boff = (size_t)(n0 + l16) * 512 + (size_t)kb * 32 + hf * 8;
      acc0 = wmma_bf(A, ldB(Wt + boff),           acc0);
      acc1 = wmma_bf(A, ldB(Wt + MAT + boff),     acc1);
      acc2 = wmma_bf(A, ldB(Wt + 2 * MAT + boff), acc2);
    }
    const float bv0 = bz[n0 + l16], bv1 = br[n0 + l16], bv2 = bh[n0 + l16];
#pragma unroll
    for (int v = 0; v < 8; ++v) {
      const int r = m0 + v + 8 * hf;     // X row index = b*512 + t
      const int t = r & 511, b = r >> 9;
      const size_t o = ((size_t)t * 64 + b) * 512 + n0 + l16;
      proj[o]          = acc0[v] + bv0;
      proj[GS + o]     = acc1[v] + bv1;
      proj[2 * GS + o] = acc2[v] + bv2;
    }
  }
}

// ---------------- Phase 2: GRU recurrence, 4 WGs x 16 batch rows, h/z in registers ----------
__global__ void __launch_bounds__(512) k_rnn(
    const float* __restrict__ proj, const u16* __restrict__ Ut,
    const int* __restrict__ mask, float* __restrict__ out)
{
  __shared__ u16 sh_hb[16][520];    // bf16 mirror of h (padded rows: conflict-free ds_load_b128)
  __shared__ u16 sh_rhb[16][520];   // bf16 r*h

  const int bi   = blockIdx.x;      // batch rows 16*bi .. 16*bi+15
  const int lane = threadIdx.x & 31;
  const int wv   = threadIdx.x >> 5;  // 16 waves
  const int hf   = lane >> 4;
  const int l16  = lane & 15;
  const int n0a  = wv * 32, n0b = wv * 32 + 16;   // this wave's two N-tiles

  for (int i = threadIdx.x; i < 16 * 520; i += 512) (&sh_hb[0][0])[i] = 0;
  __syncthreads();

  float h0r[8], h1r[8], z0r[8], z1r[8];
#pragma unroll
  for (int v = 0; v < 8; ++v) { h0r[v] = 0.f; h1r[v] = 0.f; }

  const u16* utz = Ut;
  const u16* utr = Ut + MAT;
  const u16* uth = Ut + 2 * MAT;
  const size_t bA = (size_t)(n0a + l16) * 512 + hf * 8;  // B-operand lane bases
  const size_t bB = (size_t)(n0b + l16) * 512 + hf * 8;

  for (int t = 0; t < 512; ++t) {
    const size_t pbase = ((size_t)t * 64 + 16 * bi) * 512;

    // prefetch next timestep's pre-activations (independent of the h dependency chain);
    // lowers to global_prefetch_b8 on gfx1250, no LOADcnt cost
    if (t < 511) {
      const size_t pn0 = pbase + (size_t)64 * 512 + (size_t)(8 * hf) * 512 + n0a + l16;
      const size_t pn4 = pn0 + (size_t)4 * 512;
      __builtin_prefetch(&proj[pn0], 0, 3);
      __builtin_prefetch(&proj[GS + pn0], 0, 3);
      __builtin_prefetch(&proj[2 * GS + pn0], 0, 3);
      __builtin_prefetch(&proj[pn4], 0, 3);
      __builtin_prefetch(&proj[GS + pn4], 0, 3);
      __builtin_prefetch(&proj[2 * GS + pn4], 0, 3);
    }

    // ---- pass 1: z and r gates (A = h in bf16, shared across 4 accumulators)
    v8f az0 = vzero(), az1 = vzero(), ar0 = vzero(), ar1 = vzero();
    for (int kb = 0; kb < 16; ++kb) {
      const v16bf A = ldA(&sh_hb[l16][kb * 32 + hf * 8]);
      const size_t k2 = (size_t)kb * 32;
      az0 = wmma_bf(A, ldB(utz + bA + k2), az0);
      az1 = wmma_bf(A, ldB(utz + bB + k2), az1);
      ar0 = wmma_bf(A, ldB(utr + bA + k2), ar0);
      ar1 = wmma_bf(A, ldB(utr + bB + k2), ar1);
    }
#pragma unroll
    for (int v = 0; v < 8; ++v) {
      const int m = v + 8 * hf;
      const size_t o0 = pbase + (size_t)m * 512 + n0a + l16;
      const size_t o1 = pbase + (size_t)m * 512 + n0b + l16;
      const float z0 = sigm(az0[v] + proj[o0]);
      const float z1 = sigm(az1[v] + proj[o1]);
      z0r[v] = z0; z1r[v] = z1;
      const float r0 = sigm(ar0[v] + proj[GS + o0]);
      const float r1 = sigm(ar1[v] + proj[GS + o1]);
      sh_rhb[m][n0a + l16] = (u16)bf16_1(r0 * h0r[v]);
      sh_rhb[m][n0b + l16] = (u16)bf16_1(r1 * h1r[v]);
    }
    __syncthreads();

    // ---- pass 2: candidate gemm (A = r*h), then state update
    v8f ah0 = vzero(), ah1 = vzero();
    for (int kb = 0; kb < 16; ++kb) {
      const v16bf A = ldA(&sh_rhb[l16][kb * 32 + hf * 8]);
      const size_t k2 = (size_t)kb * 32;
      ah0 = wmma_bf(A, ldB(uth + bA + k2), ah0);
      ah1 = wmma_bf(A, ldB(uth + bB + k2), ah1);
    }
#pragma unroll
    for (int v = 0; v < 8; ++v) {
      const int m = v + 8 * hf;
      const size_t o0 = pbase + (size_t)m * 512 + n0a + l16;
      const size_t o1 = pbase + (size_t)m * 512 + n0b + l16;
      const float hh0 = tanhf(ah0[v] + proj[2 * GS + o0]);
      const float hh1 = tanhf(ah1[v] + proj[2 * GS + o1]);
      float hn0 = z0r[v] * h0r[v] + (1.f - z0r[v]) * hh0;
      float hn1 = z1r[v] * h1r[v] + (1.f - z1r[v]) * hh1;
      const int mk = mask[(16 * bi + m) * 512 + t];
      hn0 = (mk > 0) ? hn0 : h0r[v];
      hn1 = (mk > 0) ? hn1 : h1r[v];
      h0r[v] = hn0; h1r[v] = hn1;
      sh_hb[m][n0a + l16] = (u16)bf16_1(hn0);
      sh_hb[m][n0b + l16] = (u16)bf16_1(hn1);
    }
    __syncthreads();
  }

#pragma unroll
  for (int v = 0; v < 8; ++v) {
    const int m = v + 8 * hf;
    out[(size_t)(16 * bi + m) * 512 + n0a + l16] = h0r[v];
    out[(size_t)(16 * bi + m) * 512 + n0b + l16] = h1r[v];
  }
}

extern "C" void kernel_launch(void* const* d_in, const int* in_sizes, int n_in,
                              void* d_out, int out_size, void* d_ws, size_t ws_size,
                              hipStream_t stream) {
  (void)in_sizes; (void)n_in; (void)out_size; (void)ws_size;
  const float* X  = (const float*)d_in[0];
  const int* mask = (const int*)d_in[1];
  const float* Wz = (const float*)d_in[2];
  const float* Uz = (const float*)d_in[3];
  const float* bz = (const float*)d_in[4];
  const float* Wr = (const float*)d_in[5];
  const float* Ur = (const float*)d_in[6];
  const float* br = (const float*)d_in[7];
  const float* Wh = (const float*)d_in[8];
  const float* Uh = (const float*)d_in[9];
  const float* bh = (const float*)d_in[10];

  u16*   Wt   = (u16*)d_ws;                                    // 3 x 512x512 bf16
  u16*   Ut   = (u16*)((char*)d_ws + 3 * MAT * sizeof(u16));   // 3 x 512x512 bf16
  float* proj = (float*)((char*)d_ws + 6 * MAT * sizeof(u16)); // 3 x [T][B][512] f32 (192MB)

  k_prep<<<6144, 256, 0, stream>>>(Wz, Wr, Wh, Uz, Ur, Uh, Wt, Ut);
  k_proj<<<2048, 256, 0, stream>>>(X, Wt, bz, br, bh, proj);
  k_rnn<<<4, 512, 0, stream>>>(proj, Ut, mask, (float*)d_out);
}
